// SimOTAssigner_34978213658777
// MI455X (gfx1250) — matrix operand order repaired and checked by reading, not compiled, not run on hardware
//
#include <hip/hip_runtime.h>
#include <hip/hip_bf16.h>

// ---------------- problem constants ----------------
#define P_ANCH   100000
#define NGT      256
#define NCLS     80
#define DSTRIDE  96          // classes padded to 3 x K=32 WMMA steps
#define IMG_WH   1280.0f
#define BBOX_B   3.0f
#define INF_C    1.0e6f
#define TAIL_C   3.0e6f      // > any real cost (cls_loss <= ~8100, +INF mask)
#define ANCH_PER_BLOCK 256   // 4 sub-tiles of 64 anchors
#define NBLK     391         // ceil(100000/256)
#define PARTK    (NBLK * 10)

typedef _Float16 v16h __attribute__((ext_vector_type(16)));
typedef float    v8f  __attribute__((ext_vector_type(8)));

// fast reciprocal (v_rcp_f32): denominators here are positive & well-scaled
__device__ __forceinline__ float fastrcp(float x) { return __builtin_amdgcn_rcpf(x); }

// ---------------- top-k helpers (lexicographic (val,idx), ascending) -------
__device__ __forceinline__ void topk_insert_min(float* vals, int* idxs, float v, int i) {
    if (v > vals[9] || (v == vals[9] && i >= idxs[9])) return;
    int j = 9;
    while (j > 0 && (v < vals[j-1] || (v == vals[j-1] && i < idxs[j-1]))) {
        vals[j] = vals[j-1]; idxs[j] = idxs[j-1]; --j;
    }
    vals[j] = v; idxs[j] = i;
}
__device__ __forceinline__ void topk_insert_max(float* vals, float v) {
    if (v <= vals[9]) return;
    int j = 9;
    while (j > 0 && v > vals[j-1]) { vals[j] = vals[j-1]; --j; }
    vals[j] = v;
}

// ---------------- kernel 0: GT preprocessing (256 GTs) ----------------
// gtT[m*8] = {x1, y1, cx, cy, x2, y2, area_mut, pad}; cg[m] = class-1
__global__ void gt_prep_kernel(const float* __restrict__ bbox_labels,
                               const int*   __restrict__ cls_labels,
                               float* __restrict__ gtT, int* __restrict__ cg) {
    int m = threadIdx.x;
    float x1 = bbox_labels[m*4+0] * IMG_WH;
    float y1 = bbox_labels[m*4+1] * IMG_WH;
    float x2 = bbox_labels[m*4+2] * IMG_WH;
    float y2 = bbox_labels[m*4+3] * IMG_WH;
    float cx = x1 + (x2 - x1) * 0.5f;
    float cy = y1 + (y2 - y1) * 0.5f;
    float area = fmaxf(x2 - cx, 0.0f) * fmaxf(y2 - cy, 0.0f); // area of bl_mut
    float* g = gtT + m*8;
    g[0]=x1; g[1]=y1; g[2]=cx; g[3]=cy; g[4]=x2; g[5]=y2; g[6]=area; g[7]=0.0f;
    cg[m] = cls_labels[m] - 1;
}

// ---------------- kernel 1: per-anchor class terms (memory-bound stream) ----
// D[n*96+c] = f16(l1p - lp), S[n] = sum_c l1p ; also (re)init match counters
__global__ __launch_bounds__(256)
void anchor_prep_kernel(const float* __restrict__ cls_preds,
                        const float* __restrict__ obj_preds,
                        _Float16* __restrict__ Dmat, float* __restrict__ Svec,
                        int* __restrict__ cnt, int* __restrict__ chosen) {
    int n = blockIdx.x * blockDim.x + threadIdx.x;
    if (n >= P_ANCH) return;
    float so = fastrcp(1.0f + __expf(-obj_preds[n]));
    const float* row = cls_preds + (size_t)n * NCLS;
    _Float16* drow = Dmat + (size_t)n * DSTRIDE;
    float s = 0.0f;
    #pragma unroll 4
    for (int c = 0; c < NCLS; ++c) {
        float sc  = fastrcp(1.0f + __expf(-row[c]));
        float p   = sqrtf(sc * so);
        float lp  = fmaxf(__logf(p), -100.0f);
        float l1p = fmaxf(__logf(1.0f - p), -100.0f);
        s += l1p;
        drow[c] = (_Float16)(l1p - lp);
    }
    #pragma unroll
    for (int c = NCLS; c < DSTRIDE; ++c) drow[c] = (_Float16)0.0f;
    Svec[n]   = s;
    cnt[n]    = 0;
    chosen[n] = 0x7fffffff;
}

// ---------------- kernel 2: fused cost tile + in-LDS reductions ------------
// Block: 128 threads (4 wave32). Covers 256 anchors (4 subs x 64) x all 256 GTs.
// One-hot A tiles (class gather) are built ONCE per block into LDS (48 KB) --
// they depend only on (t, ks, lane). Inner loop per m-tile is then just
// 3 x ds_load(v16h) + 3 chained v_wmma_f32_16x16x32_f16 + IoU/fg VALU math.
__global__ __launch_bounds__(128)
void cost_kernel(const float* __restrict__ bbox_preds,
                 const float* __restrict__ grids,
                 const _Float16* __restrict__ Dmat,
                 const float* __restrict__ Svec,
                 const float* __restrict__ gtT,
                 const int*   __restrict__ cg,
                 int*   __restrict__ argminIdx,
                 float* __restrict__ costPV, int* __restrict__ costPI,
                 float* __restrict__ iouPV) {
    __shared__ float    sCost[256 * 64];     // 64 KB  cost tile (m x local-n)
    __shared__ float    sGT[256 * 8];        //  8 KB  gt table
    __shared__ int      sCG[256];            //  1 KB  gt classes
    __shared__ float    sPB[64 * 5];         //  pred boxes of current sub
    __shared__ float    sTopV[256 * 10];     // running per-gt top-10 cost vals
    __shared__ int      sTopI[256 * 10];     // running per-gt top-10 cost idxs
    __shared__ float    sIouT[256 * 10];     // running per-gt top-10 iou vals
    __shared__ _Float16 sA[16 * 3 * 32 * 16];// 48 KB  one-hot A tiles [t][ks][lane][e]

    const int tid  = threadIdx.x;
    const int lane = tid & 31;
    const int w    = tid >> 5;     // wave 0..3
    const int col  = lane & 15;    // N within 16-wide tile
    const int hi   = lane >> 4;    // lane half (WMMA layout)

    for (int k = tid; k < 256*8; k += 128) sGT[k] = gtT[k];
    for (int k = tid; k < 256;   k += 128) sCG[k] = cg[k];
    for (int k = tid; k < 2560;  k += 128) { sTopV[k] = 1e30f; sTopI[k] = 0x7fffffff; sIouT[k] = -1e30f; }
    __syncthreads();

    // Build all 48 one-hot A tiles (16 m-tiles x 3 k-steps) once per block.
    // Layout used here (contiguous k, same as B): lane holds row M = lane&15;
    // element e <-> K = e + 16*(lane>>4) + 32*ks.
    for (int idx = tid; idx < 16*3*32; idx += 128) {
        const int tileIdx = idx >> 5;          // t*3 + ks
        const int ln      = idx & 31;
        const int t  = tileIdx / 3, ks = tileIdx % 3;
        const int c  = ln & 15,     h  = ln >> 4;
        const int scg = sCG[t*16 + c];
        _Float16* ap = &sA[(size_t)idx * 16];
        #pragma unroll
        for (int e = 0; e < 16; ++e) {
            const int K = e + h * 16 + ks * 32;
            ap[e] = (scg == K) ? (_Float16)1.0f : (_Float16)0.0f;
        }
    }
    __syncthreads();

    for (int sub = 0; sub < 4; ++sub) {
        const int nbase = blockIdx.x * ANCH_PER_BLOCK + sub * 64;
        const int n     = nbase + 16 * w + col;
        const bool valid = (n < P_ANCH);
        const int nc    = valid ? n : 0;

        float px1 = 0.f, py1 = 0.f, px2 = 0.f, py2 = 0.f, ap = 0.f;
        float gx = -1e9f, gy = -1e9f, Sn = 0.f;
        if (valid) {
            px1 = bbox_preds[(size_t)n*4+0]; py1 = bbox_preds[(size_t)n*4+1];
            px2 = bbox_preds[(size_t)n*4+2]; py2 = bbox_preds[(size_t)n*4+3];
            ap  = fmaxf(px2-px1, 0.f) * fmaxf(py2-py1, 0.f);
            gx  = grids[(size_t)n*4+0];  gy = grids[(size_t)n*4+1];
            Sn  = Svec[n];
        }
        if (hi == 0) {
            float* pb = &sPB[(16*w + col) * 5];
            pb[0]=px1; pb[1]=py1; pb[2]=px2; pb[3]=py2; pb[4]=ap;
        }

        // B tiles (K x 16 f16): element e <-> class 32*ks + 16*hi + e
        const _Float16* drow = Dmat + (size_t)nc * DSTRIDE + 16 * hi;
        v16h b0 = *(const v16h*)(drow);
        v16h b1 = *(const v16h*)(drow + 32);
        v16h b2 = *(const v16h*)(drow + 64);

        // prefetch next sub-tile's streams into cache (global_prefetch_b8)
        if (sub < 3) {
            const int np = n + 64;
            if (np < P_ANCH) {
                __builtin_prefetch(Dmat + (size_t)np * DSTRIDE, 0, 3);
                __builtin_prefetch(bbox_preds + (size_t)np * 4, 0, 3);
                __builtin_prefetch(grids + (size_t)np * 4, 0, 3);
            }
        }

        for (int t = 0; t < 16; ++t) {
            const _Float16* aBase = &sA[(size_t)((t*3)*32 + lane) * 16];
            v16h a0 = *(const v16h*)(aBase);
            v16h a1 = *(const v16h*)(aBase + 512);   // next ks tile: +32 lanes*16
            v16h a2 = *(const v16h*)(aBase + 1024);
            v8f acc = {0.f,0.f,0.f,0.f,0.f,0.f,0.f,0.f};
            acc = __builtin_amdgcn_wmma_f32_16x16x32_f16(false, a0, false, b0, (short)0, acc, false, false);
            acc = __builtin_amdgcn_wmma_f32_16x16x32_f16(false, a1, false, b1, (short)0, acc, false, false);
            acc = __builtin_amdgcn_wmma_f32_16x16x32_f16(false, a2, false, b2, (short)0, acc, false, false);
            // C layout: lane L holds N=col, VGPR r <-> M = 16t + 8*hi + r
            #pragma unroll
            for (int r = 0; r < 8; ++r) {
                const int m = t*16 + hi*8 + r;
                const float* g = &sGT[m*8];
                float c;
                if (valid) {
                    bool fg = (gx >= g[0]) & (gx <= g[4]) & (gy >= g[1]) & (gy <= g[5]);
                    float ix1 = fmaxf(g[2], px1), iy1 = fmaxf(g[3], py1);
                    float ix2 = fminf(g[4], px2), iy2 = fminf(g[5], py2);
                    float inter = fmaxf(ix2-ix1, 0.f) * fmaxf(iy2-iy1, 0.f);
                    float iou = inter * fastrcp(g[6] + ap - inter + 1e-9f);
                    c = acc[r] - Sn + BBOX_B * iou + (fg ? 0.0f : INF_C);
                } else {
                    c = TAIL_C;
                }
                sCost[m*64 + 16*w + col] = c;
            }
        }
        __syncthreads();

        // phase 2a: per-anchor argmin over all 256 GTs (tie -> lowest m)
        if (tid < 64) {
            const int na = nbase + tid;
            if (na < P_ANCH) {
                float bv = sCost[tid]; int bm = 0;
                for (int m = 1; m < 256; ++m) {
                    float v = sCost[m*64 + tid];
                    if (v < bv) { bv = v; bm = m; }
                }
                argminIdx[na] = bm;
            }
        }
        // phase 2b: per-GT running top-10 (cost asc, iou desc); 2 GTs/thread
        #pragma unroll
        for (int gg = 0; gg < 2; ++gg) {
            const int m = tid + 128 * gg;
            const float* g = &sGT[m*8];
            float* tv = &sTopV[m*10]; int* ti = &sTopI[m*10]; float* uv = &sIouT[m*10];
            for (int c2 = 0; c2 < 64; ++c2) {
                const int na = nbase + c2;
                topk_insert_min(tv, ti, sCost[m*64 + c2], na);
                const float* pb = &sPB[c2*5];
                float ix1 = fmaxf(g[2], pb[0]), iy1 = fmaxf(g[3], pb[1]);
                float ix2 = fminf(g[4], pb[2]), iy2 = fminf(g[5], pb[3]);
                float inter = fmaxf(ix2-ix1, 0.f) * fmaxf(iy2-iy1, 0.f);
                float iou = inter * fastrcp(g[6] + pb[4] - inter + 1e-9f);
                topk_insert_max(uv, iou);
            }
        }
        __syncthreads();
    }

    // write per-block partial top-10s
    #pragma unroll
    for (int gg = 0; gg < 2; ++gg) {
        const int m = tid + 128 * gg;
        const size_t base = (size_t)m * PARTK + (size_t)blockIdx.x * 10;
        #pragma unroll
        for (int j = 0; j < 10; ++j) {
            costPV[base + j] = sTopV[m*10 + j];
            costPI[base + j] = sTopI[m*10 + j];
            iouPV [base + j] = sIouT[m*10 + j];
        }
    }
}

// ---------------- kernel 3: per-GT merge of partials -> dynamic_k + picks ---
__global__ __launch_bounds__(128)
void merge_kernel(const float* __restrict__ costPV, const int* __restrict__ costPI,
                  const float* __restrict__ iouPV,
                  int* __restrict__ selIdx, int* __restrict__ selCnt) {
    __shared__ float mv[128*10]; __shared__ int mi[128*10]; __shared__ float mu[128*10];
    const int m = blockIdx.x, tid = threadIdx.x;
    float tv[10]; int ti[10]; float uv[10];
    #pragma unroll
    for (int j = 0; j < 10; ++j) { tv[j] = 1e30f; ti[j] = 0x7fffffff; uv[j] = -1e30f; }
    const size_t base = (size_t)m * PARTK;
    for (int e = tid; e < PARTK; e += 128) {
        topk_insert_min(tv, ti, costPV[base+e], costPI[base+e]);
        topk_insert_max(uv, iouPV[base+e]);
    }
    #pragma unroll
    for (int j = 0; j < 10; ++j) { mv[tid*10+j]=tv[j]; mi[tid*10+j]=ti[j]; mu[tid*10+j]=uv[j]; }
    __syncthreads();
    if (tid == 0) {
        float fv[10]; int fi[10]; float fu[10];
        #pragma unroll
        for (int j = 0; j < 10; ++j) { fv[j]=1e30f; fi[j]=0x7fffffff; fu[j]=-1e30f; }
        for (int e = 0; e < 1280; ++e) { topk_insert_min(fv, fi, mv[e], mi[e]); topk_insert_max(fu, mu[e]); }
        float s = 0.f;
        #pragma unroll
        for (int j = 0; j < 10; ++j) s += fmaxf(fu[j], 0.0f);
        int k = (int)s;               // truncation toward zero, like astype(int32)
        if (k < 1) k = 1; if (k > 10) k = 10;
        selCnt[m] = k;
        #pragma unroll
        for (int j = 0; j < 10; ++j) selIdx[m*10 + j] = fi[j];
    }
}

// ---------------- kernel 4: scatter matches (counts + unique owner) --------
__global__ void scatter_kernel(const int* __restrict__ selIdx, const int* __restrict__ selCnt,
                               int* __restrict__ cnt, int* __restrict__ chosen) {
    int m = threadIdx.x;
    int k = selCnt[m];
    for (int j = 0; j < k; ++j) {
        int n = selIdx[m*10 + j];
        if ((unsigned)n < (unsigned)P_ANCH) {
            atomicAdd(&cnt[n], 1);
            atomicMin(&chosen[n], m);
        }
    }
}

// ---------------- kernel 5: finalize per-anchor outputs --------------------
__global__ __launch_bounds__(256)
void finalize_kernel(const int* __restrict__ cnt, const int* __restrict__ chosen,
                     const int* __restrict__ argminIdx, const float* __restrict__ gtT,
                     const int* __restrict__ cls_labels, float* __restrict__ out) {
    int n = blockIdx.x * blockDim.x + threadIdx.x;
    if (n >= P_ANCH) return;
    const int c = cnt[n];
    int g = 0; float fg = 0.0f;
    if (c == 1)      { g = chosen[n];    fg = 1.0f; }
    else if (c > 1)  { g = argminIdx[n]; fg = 1.0f; }   // cost argmin resolves multi
    out[n]                 = (float)(cls_labels[g] - 1);   // assigned_cls
    out[P_ANCH + n]        = 1.0f;                         // assigned_obj
    const float* gt = gtT + g*8;                           // bl_mut = {cx,cy,x2,y2}
    out[2*P_ANCH + n*4 + 0] = gt[2];
    out[2*P_ANCH + n*4 + 1] = gt[3];
    out[2*P_ANCH + n*4 + 2] = gt[4];
    out[2*P_ANCH + n*4 + 3] = gt[5];
    out[6*P_ANCH + n]      = (float)g;                     // matched_gt_idxs
    out[7*P_ANCH + n]      = fg;                           // matched_fg_mask
}

// ---------------- host launcher --------------------------------------------
static inline size_t alignUp(size_t x) { return (x + 255) & ~(size_t)255; }

extern "C" void kernel_launch(void* const* d_in, const int* in_sizes, int n_in,
                              void* d_out, int out_size, void* d_ws, size_t ws_size,
                              hipStream_t stream) {
    const float* cls_preds   = (const float*)d_in[0];
    const float* obj_preds   = (const float*)d_in[1];
    const float* bbox_preds  = (const float*)d_in[2];
    const float* grids       = (const float*)d_in[3];
    const int*   cls_labels  = (const int*)  d_in[4];
    const float* bbox_labels = (const float*)d_in[5];
    float* out = (float*)d_out;

    // workspace carve-up (~33 MB)
    char* w = (char*)d_ws;
    size_t off = 0;
    _Float16* Dmat   = (_Float16*)(w + off); off = alignUp(off + (size_t)P_ANCH * DSTRIDE * sizeof(_Float16));
    float* Svec      = (float*)(w + off);    off = alignUp(off + (size_t)P_ANCH * sizeof(float));
    float* gtT       = (float*)(w + off);    off = alignUp(off + (size_t)NGT * 8 * sizeof(float));
    int*   cg        = (int*)(w + off);      off = alignUp(off + (size_t)NGT * sizeof(int));
    int*   argminIdx = (int*)(w + off);      off = alignUp(off + (size_t)P_ANCH * sizeof(int));
    int*   cnt       = (int*)(w + off);      off = alignUp(off + (size_t)P_ANCH * sizeof(int));
    int*   chosen    = (int*)(w + off);      off = alignUp(off + (size_t)P_ANCH * sizeof(int));
    float* costPV    = (float*)(w + off);    off = alignUp(off + (size_t)NGT * PARTK * sizeof(float));
    int*   costPI    = (int*)(w + off);      off = alignUp(off + (size_t)NGT * PARTK * sizeof(int));
    float* iouPV     = (float*)(w + off);    off = alignUp(off + (size_t)NGT * PARTK * sizeof(float));
    int*   selIdx    = (int*)(w + off);      off = alignUp(off + (size_t)NGT * 10 * sizeof(int));
    int*   selCnt    = (int*)(w + off);      off = alignUp(off + (size_t)NGT * sizeof(int));
    (void)ws_size; (void)in_sizes; (void)n_in; (void)out_size;

    gt_prep_kernel<<<1, NGT, 0, stream>>>(bbox_labels, cls_labels, gtT, cg);
    anchor_prep_kernel<<<(P_ANCH + 255) / 256, 256, 0, stream>>>(
        cls_preds, obj_preds, Dmat, Svec, cnt, chosen);
    cost_kernel<<<NBLK, 128, 0, stream>>>(
        bbox_preds, grids, Dmat, Svec, gtT, cg, argminIdx, costPV, costPI, iouPV);
    merge_kernel<<<NGT, 128, 0, stream>>>(costPV, costPI, iouPV, selIdx, selCnt);
    scatter_kernel<<<1, NGT, 0, stream>>>(selIdx, selCnt, cnt, chosen);
    finalize_kernel<<<(P_ANCH + 255) / 256, 256, 0, stream>>>(
        cnt, chosen, argminIdx, gtT, cls_labels, out);
}
